// GraphUpdater_45621142618197
// MI455X (gfx1250) — matrix operand order repaired and checked by reading, not compiled, not run on hardware
//
#include <hip/hip_runtime.h>
#include <math.h>

#define B_  32
#define R_  10
#define N_  512
#define DN_ 128
#define H_  128
#define Q_  256
#define NB_ 16
#define KCAT 256    // DN + DR
#define WBK  2560   // R * (DN+DR)

typedef _Float16 h16;
typedef __attribute__((ext_vector_type(16))) _Float16 v16h;
typedef __attribute__((ext_vector_type(8)))  _Float16 h8;
typedef __attribute__((ext_vector_type(8)))  float    v8f;

// ---- WMMA fragment helpers (CDNA5 f16 16x16x32 layouts, wave32) ----
// A (16M x 32K): lane l -> row l&15; halves = K [(l>>4)*8 .. +7] and [+16 .. +23]
__device__ __forceinline__ v16h load_a_frag(const h16* __restrict__ rowk, int kgrp) {
  h8 lo = *(const h8*)(rowk + kgrp);
  h8 hi = *(const h8*)(rowk + kgrp + 16);
  v16h a;
#pragma unroll
  for (int t = 0; t < 8; ++t) { a[t] = lo[t]; a[8 + t] = hi[t]; }
  return a;
}
// B (32K x 16N) from Bt stored (N x K) row-major: lane l -> col l&15; K [(l>>4)*8 .. +7] and [+16 .. +23]
__device__ __forceinline__ v16h load_b_frag(const h16* __restrict__ colk) {
  h8 lo = *(const h8*)(colk);
  h8 hi = *(const h8*)(colk + 16);
  v16h b;
#pragma unroll
  for (int t = 0; t < 8; ++t) { b[t] = lo[t]; b[8 + t] = hi[t]; }
  return b;
}

// ---- f32 -> f16 convert (+ optional transposed copy) ----
__global__ void k_cvt(const float* __restrict__ src, h16* __restrict__ dstN,
                      h16* __restrict__ dstT, int batch, int rows, int cols) {
  size_t idx = (size_t)blockIdx.x * blockDim.x + threadIdx.x;
  size_t total = (size_t)batch * rows * cols;
  if (idx >= total) return;
  int c = (int)(idx % cols);
  size_t rb = idx / cols;
  int r = (int)(rb % rows);
  int b = (int)(rb / rows);
  h16 v = (h16)src[idx];
  dstN[idx] = v;
  if (dstT) dstT[((size_t)b * cols + c) * rows + r] = v;
}

// ---- c[b,r,j] = rf[b,r,:] . Wb[j, r*256+128 : r*256+256] ----
__global__ void k_c(const float* __restrict__ rf, const float* __restrict__ Wb,
                    float* __restrict__ cvec) {
  int idx = blockIdx.x * blockDim.x + threadIdx.x;
  if (idx >= B_ * R_ * NB_) return;
  int j = idx % NB_;
  int r = (idx / NB_) % R_;
  int b = idx / (NB_ * R_);
  const float* rrow = rf + ((size_t)b * R_ + r) * DN_;
  const float* wrow = Wb + (size_t)j * WBK + r * KCAT + DN_;
  float s = 0.f;
  for (int d = 0; d < DN_; ++d) s += rrow[d] * wrow[d];
  cvec[idx] = s;
}

// ---- G2t[b,r,j,m] = nf[b,m,:] . Wb[j, r*256 : r*256+128] + c[b,r,j]  (f16) ----
__global__ __launch_bounds__(160) void k_g2(const float* __restrict__ nf,
                                            const float* __restrict__ Wb,
                                            const float* __restrict__ cvec,
                                            h16* __restrict__ g2t) {
  __shared__ float nfrow[DN_];
  const int b = blockIdx.x / N_;
  const int m = blockIdx.x % N_;
  const int t = threadIdx.x;
  if (t < DN_) nfrow[t] = nf[((size_t)b * N_ + m) * DN_ + t];
  __syncthreads();
  const int r = t / NB_, j = t % NB_;
  const float* wrow = Wb + (size_t)j * WBK + r * KCAT;
  float s = cvec[(b * R_ + r) * NB_ + j];
  for (int d = 0; d < DN_; ++d) s += nfrow[d] * wrow[d];
  g2t[((size_t)(b * R_ + r) * NB_ + j) * N_ + m] = (h16)s;
}

// ---- RGCN main GEMM: h = sum_r adj(b,r) @ G2(b,r) ; then sigmoid(h @ Ww^T + bias) ----
__global__ __launch_bounds__(32) void k_rgcn(const float* __restrict__ adj,
                                             const h16* __restrict__ g2t,
                                             const float* __restrict__ Ww,
                                             const float* __restrict__ rbias,
                                             h16* __restrict__ nr16,
                                             h16* __restrict__ nr16T) {
  const int b = blockIdx.y, ntile = blockIdx.x, l = threadIdx.x;
  const int row = l & 15, hi = l >> 4;
  const int kgrp = hi * 8, kbase = hi * 16;
  v8f acc = {};
  for (int r = 0; r < R_; ++r) {
    const float* adjp = adj + ((size_t)(b * R_ + r) * N_ + ntile * 16 + row) * N_;
    const h16* g2p = g2t + ((size_t)(b * R_ + r) * NB_ + row) * N_;
    for (int kc = 0; kc < N_; kc += 32) {
      float4 f0 = *(const float4*)(adjp + kc + kgrp);
      float4 f1 = *(const float4*)(adjp + kc + kgrp + 4);
      float4 f2 = *(const float4*)(adjp + kc + kgrp + 16);
      float4 f3 = *(const float4*)(adjp + kc + kgrp + 20);
      if (kc + 32 < N_) __builtin_prefetch(adjp + kc + 32 + kgrp, 0, 1);
      v16h a;
      a[0] = (h16)f0.x; a[1] = (h16)f0.y; a[2]  = (h16)f0.z; a[3]  = (h16)f0.w;
      a[4] = (h16)f1.x; a[5] = (h16)f1.y; a[6]  = (h16)f1.z; a[7]  = (h16)f1.w;
      a[8] = (h16)f2.x; a[9] = (h16)f2.y; a[10] = (h16)f2.z; a[11] = (h16)f2.w;
      a[12] = (h16)f3.x; a[13] = (h16)f3.y; a[14] = (h16)f3.z; a[15] = (h16)f3.w;
      v16h bb = load_b_frag(g2p + kc + kgrp);
      acc = __builtin_amdgcn_wmma_f32_16x16x32_f16(false, a, false, bb, (short)0,
                                                   acc, false, false);
    }
  }
  __shared__ float hbuf[256];
#pragma unroll
  for (int g = 0; g < 8; ++g) hbuf[(g + 8 * hi) * 16 + row] = acc[g];
  __syncthreads();
  const float bias = rbias[0];
  for (int idx = l; idx < 16 * H_; idx += 32) {
    int rr = idx >> 7;
    int hc = idx & (H_ - 1);
    float z = bias;
#pragma unroll
    for (int j = 0; j < NB_; ++j) z += hbuf[rr * 16 + j] * Ww[hc * NB_ + j];
    float s = 1.0f / (1.0f + __expf(-z));
    int n = ntile * 16 + rr;
    nr16[((size_t)b * N_ + n) * H_ + hc] = (h16)s;
    nr16T[((size_t)b * H_ + hc) * N_ + n] = (h16)s;
  }
}

// ---- row-wise dots with w_C / w_Q ----
__global__ void k_rvec(const h16* __restrict__ x16, const float* __restrict__ wC,
                       const float* __restrict__ wQ, float* __restrict__ outC,
                       float* __restrict__ outQ, int totalRows) {
  int idx = blockIdx.x * blockDim.x + threadIdx.x;
  if (idx >= totalRows) return;
  const h16* rowp = x16 + (size_t)idx * H_;
  float sc = 0.f, sq = 0.f;
  for (int h = 0; h < H_; ++h) {
    float v = (float)rowp[h];
    sc += v * wC[h];
    sq += v * wQ[h];
  }
  outC[idx] = sc;
  outQ[idx] = sq;
}

// ---- S = (ctx * w_CQ) @ query^T + rC + rQ + bias ----
__global__ __launch_bounds__(32) void k_S(const h16* __restrict__ ctx16,
                                          const h16* __restrict__ q16,
                                          const float* __restrict__ wCQ,
                                          const float* __restrict__ rC,
                                          const float* __restrict__ rQ,
                                          const float* __restrict__ cqb,
                                          float* __restrict__ S, int Lc, int Lq) {
  const int b = blockIdx.z, ct = blockIdx.y, qt = blockIdx.x, l = threadIdx.x;
  const int lane15 = l & 15, hi = l >> 4, kgrp = hi * 8, kbase = hi * 16;
  const h16* arow = ctx16 + ((size_t)b * Lc + ct * 16 + lane15) * H_;
  const h16* brow = q16 + ((size_t)b * Lq + qt * 16 + lane15) * H_;
  v8f acc = {};
  for (int kc = 0; kc < H_; kc += 32) {
    v16h a;
#pragma unroll
    for (int t = 0; t < 8; ++t) {
      a[t]     = arow[kc + kgrp + t]      * (h16)wCQ[kc + kgrp + t];
      a[8 + t] = arow[kc + kgrp + 16 + t] * (h16)wCQ[kc + kgrp + 16 + t];
    }
    v16h bb = load_b_frag(brow + kc + kgrp);
    acc = __builtin_amdgcn_wmma_f32_16x16x32_f16(false, a, false, bb, (short)0,
                                                 acc, false, false);
  }
  const float bias = cqb[0];
#pragma unroll
  for (int g = 0; g < 8; ++g) {
    int m = ct * 16 + g + 8 * hi;
    int n = qt * 16 + lane15;
    S[((size_t)b * Lc + m) * Lq + n] =
        acc[g] + rC[(size_t)b * Lc + m] + rQ[(size_t)b * Lq + n] + bias;
  }
}

// ---- softmax over q (last axis), masked; write f16 ----
__global__ __launch_bounds__(256) void k_smax_row(const float* __restrict__ S,
                                                  const float* __restrict__ qmask,
                                                  h16* __restrict__ sq16, int Lc, int Lq) {
  const int b = blockIdx.y, c = blockIdx.x, t = threadIdx.x;
  const float* row = S + ((size_t)b * Lc + c) * Lq;
  __shared__ float red[256];
  float mx = -__builtin_inff();
  for (int q = t; q < Lq; q += 256) {
    float v = (qmask[(size_t)b * Lq + q] > 0.f) ? row[q] : -__builtin_inff();
    mx = fmaxf(mx, v);
  }
  red[t] = mx;
  __syncthreads();
  for (int s = 128; s > 0; s >>= 1) { if (t < s) red[t] = fmaxf(red[t], red[t + s]); __syncthreads(); }
  mx = red[0];
  __syncthreads();
  float acc = 0.f;
  for (int q = t; q < Lq; q += 256) {
    float v = (qmask[(size_t)b * Lq + q] > 0.f) ? row[q] : -__builtin_inff();
    acc += (v == -__builtin_inff()) ? 0.f : __expf(v - mx);
  }
  red[t] = acc;
  __syncthreads();
  for (int s = 128; s > 0; s >>= 1) { if (t < s) red[t] += red[t + s]; __syncthreads(); }
  float inv = 1.f / red[0];
  for (int q = t; q < Lq; q += 256) {
    float v = (qmask[(size_t)b * Lq + q] > 0.f) ? row[q] : -__builtin_inff();
    float e = (v == -__builtin_inff()) ? 0.f : __expf(v - mx);
    sq16[((size_t)b * Lc + c) * Lq + q] = (h16)(e * inv);
  }
}

// ---- softmax over c (axis 1), masked; write TRANSPOSED f16 (B, Lq, Lc) ----
__global__ __launch_bounds__(256) void k_smax_col(const float* __restrict__ S,
                                                  const float* __restrict__ cmask,
                                                  h16* __restrict__ scT, int Lc, int Lq) {
  const int b = blockIdx.y, q = blockIdx.x, t = threadIdx.x;
  const float* colp = S + (size_t)b * Lc * Lq + q;
  __shared__ float red[256];
  float mx = -__builtin_inff();
  for (int c = t; c < Lc; c += 256) {
    float v = (cmask[(size_t)b * Lc + c] > 0.f) ? colp[(size_t)c * Lq] : -__builtin_inff();
    mx = fmaxf(mx, v);
  }
  red[t] = mx;
  __syncthreads();
  for (int s = 128; s > 0; s >>= 1) { if (t < s) red[t] = fmaxf(red[t], red[t + s]); __syncthreads(); }
  mx = red[0];
  __syncthreads();
  float acc = 0.f;
  for (int c = t; c < Lc; c += 256) {
    float v = (cmask[(size_t)b * Lc + c] > 0.f) ? colp[(size_t)c * Lq] : -__builtin_inff();
    acc += (v == -__builtin_inff()) ? 0.f : __expf(v - mx);
  }
  red[t] = acc;
  __syncthreads();
  for (int s = 128; s > 0; s >>= 1) { if (t < s) red[t] += red[t + s]; __syncthreads(); }
  float inv = 1.f / red[0];
  for (int c = t; c < Lc; c += 256) {
    float v = (cmask[(size_t)b * Lc + c] > 0.f) ? colp[(size_t)c * Lq] : -__builtin_inff();
    float e = (v == -__builtin_inff()) ? 0.f : __expf(v - mx);
    scT[((size_t)b * Lq + q) * Lc + c] = (h16)(e * inv);
  }
}

// ---- generic batched WMMA GEMM: D = A(MxK) @ Bt(NxK)^T ; optional normal/transposed f16 out ----
__global__ __launch_bounds__(32) void k_gemm(const h16* __restrict__ A, size_t aBatch, int lda,
                                             const h16* __restrict__ Bt, size_t bBatch,
                                             h16* __restrict__ outN, h16* __restrict__ outT,
                                             int M, int Nn, int K) {
  const int b = blockIdx.z, mt = blockIdx.y, nt = blockIdx.x, l = threadIdx.x;
  const int lane15 = l & 15, hi = l >> 4, kgrp = hi * 8, kbase = hi * 16;
  const h16* arow = A + (size_t)b * aBatch + (size_t)(mt * 16 + lane15) * lda;
  const h16* brow = Bt + (size_t)b * bBatch + (size_t)(nt * 16 + lane15) * K;
  v8f acc = {};
  for (int kc = 0; kc < K; kc += 32) {
    v16h a = load_a_frag(arow + kc, kgrp);
    v16h bb = load_b_frag(brow + kc + kgrp);
    acc = __builtin_amdgcn_wmma_f32_16x16x32_f16(false, a, false, bb, (short)0,
                                                 acc, false, false);
  }
  if (outN) {
#pragma unroll
    for (int g = 0; g < 8; ++g)
      outN[(size_t)b * M * Nn + (size_t)(mt * 16 + g + 8 * hi) * Nn + nt * 16 + lane15] =
          (h16)acc[g];
  }
  if (outT) {
    h8 v;
#pragma unroll
    for (int g = 0; g < 8; ++g) v[g] = (h16)acc[g];
    *(h8*)(outT + (size_t)b * Nn * M + (size_t)(nt * 16 + lane15) * M + mt * 16 + 8 * hi) = v;
  }
}

// ---- fused concat([ctx,P,ctx*P,ctx*Qm]) @ linW^T + lin_b -> f32 output ----
__global__ __launch_bounds__(32) void k_out(const h16* __restrict__ ctx16,
                                            const h16* __restrict__ P16,
                                            const h16* __restrict__ Qm16,
                                            const h16* __restrict__ linW16,
                                            const float* __restrict__ linb,
                                            float* __restrict__ out, int Lc) {
  const int b = blockIdx.z, mt = blockIdx.y, nt = blockIdx.x, l = threadIdx.x;
  const int lane15 = l & 15, hi = l >> 4, kgrp = hi * 8, kbase = hi * 16;
  const size_t roff = ((size_t)b * Lc + mt * 16 + lane15) * H_;
  const h16* crow = ctx16 + roff;
  const h16* prow = P16 + roff;
  const h16* qrow = Qm16 + roff;
  const h16* brow = linW16 + (size_t)(nt * 16 + lane15) * (4 * H_);
  v8f acc = {};
  for (int kc = 0; kc < 4 * H_; kc += 32) {
    int seg = kc >> 7;
    int kk = (kc & 127) + kgrp;
    h8 c0 = *(const h8*)(crow + kk);
    h8 c1 = *(const h8*)(crow + kk + 16);
    h8 lo, hv;
    if (seg == 0) { lo = c0; hv = c1; }
    else if (seg == 1) { lo = *(const h8*)(prow + kk); hv = *(const h8*)(prow + kk + 16); }
    else if (seg == 2) { lo = c0 * *(const h8*)(prow + kk); hv = c1 * *(const h8*)(prow + kk + 16); }
    else { lo = c0 * *(const h8*)(qrow + kk); hv = c1 * *(const h8*)(qrow + kk + 16); }
    v16h a;
#pragma unroll
    for (int t = 0; t < 8; ++t) { a[t] = lo[t]; a[8 + t] = hv[t]; }
    v16h bb = load_b_frag(brow + kc + kgrp);
    acc = __builtin_amdgcn_wmma_f32_16x16x32_f16(false, a, false, bb, (short)0,
                                                 acc, false, false);
  }
#pragma unroll
  for (int g = 0; g < 8; ++g) {
    int m = mt * 16 + g + 8 * hi;
    int n = nt * 16 + lane15;
    out[((size_t)b * Lc + m) * H_ + n] = acc[g] + linb[n];
  }
}

extern "C" void kernel_launch(void* const* d_in, const int* in_sizes, int n_in,
                              void* d_out, int out_size, void* d_ws, size_t ws_size,
                              hipStream_t stream) {
  const float* nf    = (const float*)d_in[0];
  const float* rf    = (const float*)d_in[1];
  const float* adj   = (const float*)d_in[2];
  const float* obs   = (const float*)d_in[3];
  const float* nmask = (const float*)d_in[4];
  const float* omask = (const float*)d_in[5];
  const float* Wb    = (const float*)d_in[6];
  const float* Ww    = (const float*)d_in[7];
  const float* rbias = (const float*)d_in[8];
  const float* wC    = (const float*)d_in[9];
  const float* wQ    = (const float*)d_in[10];
  const float* wCQ   = (const float*)d_in[11];
  const float* cqb   = (const float*)d_in[12];
  const float* linW  = (const float*)d_in[13];
  const float* linb  = (const float*)d_in[14];
  float* out = (float*)d_out;

  char* p = (char*)d_ws;
  auto alloc = [&](size_t bytes) { char* r = p; p += (bytes + 255) & ~(size_t)255; return r; };
  float* cvec  = (float*)alloc((size_t)B_ * R_ * NB_ * 4);
  h16* g2t     = (h16*)alloc((size_t)B_ * R_ * NB_ * N_ * 2);
  h16* nr16    = (h16*)alloc((size_t)B_ * N_ * H_ * 2);
  h16* nr16T   = (h16*)alloc((size_t)B_ * H_ * N_ * 2);
  h16* obs16   = (h16*)alloc((size_t)B_ * Q_ * H_ * 2);
  h16* obs16T  = (h16*)alloc((size_t)B_ * H_ * Q_ * 2);
  h16* linW16  = (h16*)alloc((size_t)H_ * 4 * H_ * 2);
  float* rObsC = (float*)alloc((size_t)B_ * Q_ * 4);
  float* rObsQ = (float*)alloc((size_t)B_ * Q_ * 4);
  float* rNodC = (float*)alloc((size_t)B_ * N_ * 4);
  float* rNodQ = (float*)alloc((size_t)B_ * N_ * 4);
  float* Sbuf  = (float*)alloc((size_t)B_ * N_ * Q_ * 4);
  h16* sq16    = (h16*)alloc((size_t)B_ * N_ * Q_ * 2);
  h16* scT16   = (h16*)alloc((size_t)B_ * N_ * Q_ * 2);
  h16* At16    = (h16*)alloc((size_t)B_ * H_ * N_ * 2);
  h16* P16     = (h16*)alloc((size_t)B_ * N_ * H_ * 2);
  h16* Qm16    = (h16*)alloc((size_t)B_ * N_ * H_ * 2);

  // f16 staging
  {
    size_t tot = (size_t)B_ * Q_ * H_;
    k_cvt<<<(unsigned)((tot + 255) / 256), 256, 0, stream>>>(obs, obs16, obs16T, B_, Q_, H_);
    size_t tot2 = (size_t)H_ * 4 * H_;
    k_cvt<<<(unsigned)((tot2 + 255) / 256), 256, 0, stream>>>(linW, linW16, nullptr, 1, H_, 4 * H_);
  }
  // RGCN
  k_c<<<(B_ * R_ * NB_ + 127) / 128, 128, 0, stream>>>(rf, Wb, cvec);
  k_g2<<<B_ * N_, 160, 0, stream>>>(nf, Wb, cvec, g2t);
  k_rgcn<<<dim3(N_ / 16, B_), 32, 0, stream>>>(adj, g2t, Ww, rbias, nr16, nr16T);
  // trilinear rank-1 vectors
  k_rvec<<<(B_ * Q_ + 127) / 128, 128, 0, stream>>>(obs16, wC, wQ, rObsC, rObsQ, B_ * Q_);
  k_rvec<<<(B_ * N_ + 127) / 128, 128, 0, stream>>>(nr16, wC, wQ, rNodC, rNodQ, B_ * N_);

  // ---- direction 1: ctx = obs (Lc=Q), query = node_repr (Lq=N) ----
  k_S<<<dim3(N_ / 16, Q_ / 16, B_), 32, 0, stream>>>(obs16, nr16, wCQ, rObsC, rNodQ, cqb, Sbuf, Q_, N_);
  k_smax_row<<<dim3(Q_, B_), 256, 0, stream>>>(Sbuf, nmask, sq16, Q_, N_);
  k_smax_col<<<dim3(N_, B_), 256, 0, stream>>>(Sbuf, omask, scT16, Q_, N_);
  // A = s_ctx^T @ ctx : M=N_, K=Q_, N=H  -> At16 (H, N_)
  k_gemm<<<dim3(H_ / 16, N_ / 16, B_), 32, 0, stream>>>(scT16, (size_t)N_ * Q_, Q_,
                                                        obs16T, (size_t)H_ * Q_,
                                                        nullptr, At16, N_, H_, Q_);
  // P = s_query @ query : M=Q_, K=N_, N=H
  k_gemm<<<dim3(H_ / 16, Q_ / 16, B_), 32, 0, stream>>>(sq16, (size_t)Q_ * N_, N_,
                                                        nr16T, (size_t)H_ * N_,
                                                        P16, nullptr, Q_, H_, N_);
  // Qm = s_query @ A : M=Q_, K=N_, N=H
  k_gemm<<<dim3(H_ / 16, Q_ / 16, B_), 32, 0, stream>>>(sq16, (size_t)Q_ * N_, N_,
                                                        At16, (size_t)H_ * N_,
                                                        Qm16, nullptr, Q_, H_, N_);
  k_out<<<dim3(H_ / 16, Q_ / 16, B_), 32, 0, stream>>>(obs16, P16, Qm16, linW16, linb, out, Q_);

  // ---- direction 2: ctx = node_repr (Lc=N), query = obs (Lq=Q) ----
  k_S<<<dim3(Q_ / 16, N_ / 16, B_), 32, 0, stream>>>(nr16, obs16, wCQ, rNodC, rObsQ, cqb, Sbuf, N_, Q_);
  k_smax_row<<<dim3(N_, B_), 256, 0, stream>>>(Sbuf, omask, sq16, N_, Q_);
  k_smax_col<<<dim3(Q_, B_), 256, 0, stream>>>(Sbuf, nmask, scT16, N_, Q_);
  // A = s_ctx^T @ ctx : M=Q_, K=N_, N=H -> At16 (H, Q_)
  k_gemm<<<dim3(H_ / 16, Q_ / 16, B_), 32, 0, stream>>>(scT16, (size_t)Q_ * N_, N_,
                                                        nr16T, (size_t)H_ * N_,
                                                        nullptr, At16, Q_, H_, N_);
  // P = s_query @ query : M=N_, K=Q_, N=H
  k_gemm<<<dim3(H_ / 16, N_ / 16, B_), 32, 0, stream>>>(sq16, (size_t)N_ * Q_, Q_,
                                                        obs16T, (size_t)H_ * Q_,
                                                        P16, nullptr, N_, H_, Q_);
  // Qm = s_query @ A : M=N_, K=Q_, N=H
  k_gemm<<<dim3(H_ / 16, N_ / 16, B_), 32, 0, stream>>>(sq16, (size_t)N_ * Q_, Q_,
                                                        At16, (size_t)H_ * Q_,
                                                        Qm16, nullptr, N_, H_, Q_);
  k_out<<<dim3(H_ / 16, N_ / 16, B_), 32, 0, stream>>>(nr16, P16, Qm16, linW16, linb,
                                                       out + (size_t)B_ * Q_ * H_, N_);
  (void)in_sizes; (void)n_in; (void)out_size; (void)ws_size;
}
